// RecurrentAM_3384434229668
// MI455X (gfx1250) — matrix operand order repaired and checked by reading, not compiled, not run on hardware
//
#include <hip/hip_runtime.h>
#include <hip/hip_bf16.h>
#include <math.h>

// ---------------------------------------------------------------------------
// Problem dims (from reference)
// ---------------------------------------------------------------------------
#define SEQ     256
#define BATCH   128
#define DM      512
#define NHEAD   8
#define HDIM    64
#define HIDN    1536          // GRU hidden
#define GATES   4608          // 3*HIDN
#define NWG     128           // persistent workgroups (all must be co-resident)
#define NTHR    256           // 8 wave32s per WG

// LDS weight staging: slab = 48 rows (3 gates x 16 cols) x 256 K of bf16
#define SLABK   256
#define SLABB   (48 * SLABK * 2)          // 24576 bytes per buffer
#define SLAB_XFERS (SLABB / 16)           // 1536 x b128 transfers
#define XFERS_PER_THREAD (SLAB_XFERS / NTHR)  // 6 -> 6 async ops per wave

typedef __bf16 v16bf __attribute__((ext_vector_type(16)));
typedef float  v8f   __attribute__((ext_vector_type(8)));

// ---------------------------------------------------------------------------
// Workspace layout (bytes). Weights/x as bf16 (L2-resident: ~58 MB total),
// state region (h, r, A, sync) is zeroed each launch via hipMemsetAsync.
// ---------------------------------------------------------------------------
#define OFF_WIH  ((size_t)0)                                        // bf16 [4608,1024]
#define OFF_WHH  (OFF_WIH + (size_t)GATES*1024*2)                   // bf16 [4608,1536]
#define OFF_WO   (OFF_WHH + (size_t)GATES*HIDN*2)                   // bf16 [512,512]
#define OFF_X    (OFF_WO  + (size_t)DM*DM*2)                        // bf16 [256,128,512]
#define OFF_ZERO (OFF_X   + (size_t)SEQ*BATCH*DM*2)                 // ---- zeroed region ----
#define OFF_H32  (OFF_ZERO)                                         // f32  [128,1536]
#define OFF_H16  (OFF_H32 + (size_t)BATCH*HIDN*4)                   // bf16 [128,1536]
#define OFF_R16  (OFF_H16 + (size_t)BATCH*HIDN*2)                   // bf16 [128,512]
#define OFF_A    (OFF_R16 + (size_t)BATCH*DM*2)                     // f32  [128,8,64,64]
#define OFF_SYNC (OFF_A   + (size_t)BATCH*NHEAD*HDIM*HDIM*4)        // 2x u32 + pad
#define WS_NEED  (OFF_SYNC + 256)
#define ZERO_BYTES (WS_NEED - OFF_ZERO)

// ---------------------------------------------------------------------------
// fp32 -> bf16 conversion (grid-stride)
// ---------------------------------------------------------------------------
__global__ void cvt_bf16_kernel(const float* __restrict__ s, __bf16* __restrict__ d, int n) {
    int i = blockIdx.x * blockDim.x + threadIdx.x;
    int stride = gridDim.x * blockDim.x;
    for (; i < n; i += stride) d[i] = (__bf16)s[i];
}

// ---------------------------------------------------------------------------
// WMMA fragment loaders. Layouts (wave32); A/B use this code's contiguous k order, not the ISA table order:
//  A (16x32 bf16, row-major src [M x ldk]): lane L row M=L&15; half = L>>4;
//    elems j <- K = k0 + 16*half + j (contiguous 32B = 2x16B)
//  B (32x16 bf16 from W[N][K] row-major, B(k,n)=W[n][k]): lane L col N=L&15;
//    elems j <- K = k0 + 16*half + j (contiguous 32B = 2x16B)
//  C/D (16x16 f32): lane L col N=L&15, VGPR v row M = v + 8*(L>>4)
// ---------------------------------------------------------------------------
union Frag { uint4 u[2]; v16bf v; };

__device__ __forceinline__ v16bf load_a_frag(const __bf16* __restrict__ row, int k0, int lane) {
    const int half = (lane >> 4) & 1;
    Frag f;
    f.u[0] = *reinterpret_cast<const uint4*>(row + k0 + 16 * half);
    f.u[1] = *reinterpret_cast<const uint4*>(row + k0 + 16 * half + 8);
    return f.v;
}

// B fragment from a global weight matrix W[N][K] (used by final Wo GEMM)
__device__ __forceinline__ v16bf load_b_frag(const __bf16* __restrict__ W, int ldk,
                                             int nrow, int k0, int lane) {
    const int half = (lane >> 4) & 1;
    const __bf16* p = W + (size_t)nrow * ldk + k0 + 16 * half;
    Frag f;
    f.u[0] = *reinterpret_cast<const uint4*>(p);
    f.u[1] = *reinterpret_cast<const uint4*>(p + 8);
    return f.v;
}

// B fragment from an LDS-staged slab: row-major [48 rows][SLABK] bf16,
// row = gate*16 + n.
__device__ __forceinline__ v16bf load_b_lds(const __bf16* lb, int gate, int kkl, int lane) {
    const int half = (lane >> 4) & 1;
    const __bf16* p = lb + ((gate * 16 + (lane & 15)) * SLABK) + kkl + 16 * half;
    Frag f;
    f.u[0] = *reinterpret_cast<const uint4*>(p);
    f.u[1] = *reinterpret_cast<const uint4*>(p + 8);
    return f.v;
}

__device__ __forceinline__ v8f wmma_bf16(v16bf a, v16bf b, v8f c) {
    return __builtin_amdgcn_wmma_f32_16x16x32_bf16(false, a, false, b, (short)0, c, false, false);
}

__device__ __forceinline__ float sigmoidf_(float x) { return 1.f / (1.f + __expf(-x)); }
__device__ __forceinline__ float eluf_(float x)     { return x > 0.f ? x : __expf(x) - 1.f; }

// ---------------------------------------------------------------------------
// Async-stage one 48x256 bf16 weight slab into LDS (ASYNCcnt path).
// 1536 b128 lane-transfers, 6 per thread => 6 async instructions per wave.
// All 256 threads execute uniformly (EXEC all-ones).
// ---------------------------------------------------------------------------
__device__ __forceinline__ void stage_slab(const __bf16* __restrict__ W, int ldk,
                                           int c0, int k0, unsigned lds_off) {
    #pragma unroll
    for (int i = 0; i < XFERS_PER_THREAD; ++i) {
        const int j     = threadIdx.x + i * NTHR;   // 0..1535
        const int row   = j >> 5;                   // 0..47
        const int chunk = j & 31;                   // 16B chunk within row
        const int gate  = row >> 4;
        const int n     = row & 15;
        const __bf16* src = W + (size_t)(gate * HIDN + c0 + n) * ldk + k0 + chunk * 8;
        const unsigned dst = lds_off + (unsigned)j * 16u;
        asm volatile("global_load_async_to_lds_b128 %0, %1, off"
                     :: "v"(dst), "v"((unsigned long long)src)
                     : "memory");
    }
}

__device__ __forceinline__ void wait_async_le(int more_outstanding) {
    if (more_outstanding)
        asm volatile("s_wait_asynccnt 0x6" ::: "memory");  // keep next slab in flight
    else
        asm volatile("s_wait_asynccnt 0x0" ::: "memory");
}

// ---------------------------------------------------------------------------
// Grid-wide barrier (counter + generation). All NWG blocks must participate.
// ---------------------------------------------------------------------------
__device__ __forceinline__ void grid_barrier(unsigned* cnt, unsigned* gen, unsigned expect) {
    __syncthreads();
    if (threadIdx.x == 0) {
        __threadfence();
        unsigned prev = atomicAdd(cnt, 1u);
        if (prev == (unsigned)(NWG - 1)) {
            atomicExch(cnt, 0u);
            atomicAdd(gen, 1u);                    // release this generation
        } else {
            while (atomicAdd(gen, 0u) < expect) {  // acquire-spin
                __builtin_amdgcn_s_sleep(1);
            }
        }
        __threadfence();
    }
    __syncthreads();
}

// ---------------------------------------------------------------------------
// Persistent recurrent kernel: 128 WGs x 256 threads, loops over SEQ steps.
// ---------------------------------------------------------------------------
__global__ void __launch_bounds__(NTHR)
recurrent_am_kernel(char* __restrict__ ws,
                    const float* __restrict__ bih,
                    const float* __restrict__ bhh,
                    const float* __restrict__ bo,
                    float* __restrict__ out) {
    const __bf16* wih = (const __bf16*)(ws + OFF_WIH);
    const __bf16* whh = (const __bf16*)(ws + OFF_WHH);
    const __bf16* wo  = (const __bf16*)(ws + OFF_WO);
    const __bf16* xbf = (const __bf16*)(ws + OFF_X);
    float*  h32 = (float*)(ws + OFF_H32);
    __bf16* hbf = (__bf16*)(ws + OFF_H16);
    __bf16* rbf = (__bf16*)(ws + OFF_R16);
    float*  Aws = (float*)(ws + OFF_A);
    unsigned* cnt = (unsigned*)(ws + OFF_SYNC);
    unsigned* gen = cnt + 1;

    __shared__ __align__(16) unsigned char ldsB[2 * SLABB];   // 48 KB double buffer
    const unsigned lds0 = (unsigned)(unsigned long long)(const void*)ldsB;

    const int wave = threadIdx.x >> 5;
    const int lane = threadIdx.x & 31;
    const int nlo  = lane & 15;          // col within 16-wide tile (B/C/D layout)
    const int m0   = wave * 16;          // 16-row batch tile per wave
    const int mrow = m0 + nlo;           // A-fragment row for this lane
    const int mtop = m0 + 8 * (lane >> 4);

    unsigned nbar = 0;

    for (int t = 0; t < SEQ; ++t) {
        // ================= Phase 1: GRU gate GEMMs + h update ==============
        if (blockIdx.x < HIDN / 16) {                 // 96 WGs, 16 h-cols each
            const int c0  = blockIdx.x * 16;
            const int col = c0 + nlo;

            v8f acc_r = {}, acc_z = {}, acc_in = {}, acc_hn = {};

            const __bf16* xrow = xbf + ((size_t)t * BATCH + mrow) * DM;
            const __bf16* rrow = rbf + (size_t)mrow * DM;
            const __bf16* hrow = hbf + (size_t)mrow * HIDN;

            // ---- gi = xr @ Wih.T  (K = 1024 : [x_t | r]); 4 LDS slabs -----
            {
                const int nslab = (2 * DM) / SLABK;   // 4
                stage_slab(wih, 2 * DM, c0, 0, lds0);
                for (int s = 0; s < nslab; ++s) {
                    const int buf = s & 1;
                    if (s + 1 < nslab)
                        stage_slab(wih, 2 * DM, c0, (s + 1) * SLABK, lds0 + (buf ^ 1) * SLABB);
                    wait_async_le(s + 1 < nslab);
                    __syncthreads();                   // slab s visible to all waves
                    const __bf16* lb = (const __bf16*)(ldsB + buf * SLABB);
                    __builtin_prefetch(xrow + (s + 1) * SLABK, 0, 1);
                    #pragma unroll
                    for (int kkl = 0; kkl < SLABK; kkl += 32) {
                        const int kk = s * SLABK + kkl;
                        v16bf a = (kk < DM) ? load_a_frag(xrow, kk, lane)
                                            : load_a_frag(rrow, kk - DM, lane);
                        acc_r  = wmma_bf16(a, load_b_lds(lb, 0, kkl, lane), acc_r);
                        acc_z  = wmma_bf16(a, load_b_lds(lb, 1, kkl, lane), acc_z);
                        acc_in = wmma_bf16(a, load_b_lds(lb, 2, kkl, lane), acc_in);
                    }
                    __syncthreads();                   // done reading buf
                }
            }
            // ---- gh = h @ Whh.T  (K = 1536); 6 LDS slabs ------------------
            {
                const int nslab = HIDN / SLABK;       // 6
                stage_slab(whh, HIDN, c0, 0, lds0);
                for (int s = 0; s < nslab; ++s) {
                    const int buf = s & 1;
                    if (s + 1 < nslab)
                        stage_slab(whh, HIDN, c0, (s + 1) * SLABK, lds0 + (buf ^ 1) * SLABB);
                    wait_async_le(s + 1 < nslab);
                    __syncthreads();
                    const __bf16* lb = (const __bf16*)(ldsB + buf * SLABB);
                    __builtin_prefetch(hrow + (s + 1) * SLABK, 0, 1);
                    #pragma unroll
                    for (int kkl = 0; kkl < SLABK; kkl += 32) {
                        const int kk = s * SLABK + kkl;
                        v16bf a = load_a_frag(hrow, kk, lane);
                        acc_r  = wmma_bf16(a, load_b_lds(lb, 0, kkl, lane), acc_r);  // ir+hr
                        acc_z  = wmma_bf16(a, load_b_lds(lb, 1, kkl, lane), acc_z);  // iz+hz
                        acc_hn = wmma_bf16(a, load_b_lds(lb, 2, kkl, lane), acc_hn); // hn
                    }
                    __syncthreads();
                }
            }

            const float br_ = bih[col] + bhh[col];
            const float bz_ = bih[HIDN + col] + bhh[HIDN + col];
            const float bi_ = bih[2 * HIDN + col];
            const float bh_ = bhh[2 * HIDN + col];
            #pragma unroll
            for (int v = 0; v < 8; ++v) {
                const int m = mtop + v;
                float rg   = sigmoidf_(acc_r[v] + br_);
                float zg   = sigmoidf_(acc_z[v] + bz_);
                float nn   = tanhf(acc_in[v] + bi_ + rg * (acc_hn[v] + bh_));
                float hold = h32[(size_t)m * HIDN + col];
                float hnew = (1.f - zg) * nn + zg * hold;
                h32[(size_t)m * HIDN + col] = hnew;
                hbf[(size_t)m * HIDN + col] = (__bf16)hnew;
            }
        }

        grid_barrier(cnt, gen, ++nbar);   // h complete, visible device-wide

        // ====== Phase 2: per-(batch,head) fast-weight update + readout =====
        {
            const int b    = blockIdx.x;     // 128 WGs : one batch element each
            const int head = wave;           // 8 waves : one head each
            const float* hr = h32 + (size_t)b * HIDN;
            const int d0 = 2 * lane;         // each lane owns 2 of 64 head dims

            float q0 = hr[head * HDIM + d0],          q1 = hr[head * HDIM + d0 + 1];
            float k0 = hr[DM + head * HDIM + d0],     k1 = hr[DM + head * HDIM + d0 + 1];
            float v0 = hr[2 * DM + head * HDIM + d0], v1 = hr[2 * DM + head * HDIM + d0 + 1];

            float qs = q0 * q0 + q1 * q1;
            float ks = k0 * k0 + k1 * k1;
            #pragma unroll
            for (int off = 16; off; off >>= 1) {      // wave32 reduction
                qs += __shfl_xor(qs, off, 32);
                ks += __shfl_xor(ks, off, 32);
            }
            const float qi = rsqrtf(qs), ki = rsqrtf(ks);
            q0 *= qi; q1 *= qi;
            const float ke0 = eluf_(k0 * ki), ke1 = eluf_(k1 * ki);

            float* row0 = Aws + (((size_t)b * NHEAD + head) * HDIM + d0) * HDIM;
            float* row1 = row0 + HDIM;
            float r0 = 0.f, r1 = 0.f;
            for (int qq = 0; qq < HDIM; ++qq) {       // A += k (x) v ; r = q . A
                const int src = qq >> 1;
                const float kq = (qq & 1) ? __shfl(ke1, src, 32) : __shfl(ke0, src, 32);
                const float qv = (qq & 1) ? __shfl(q1,  src, 32) : __shfl(q0,  src, 32);
                float a0 = row0[qq] + kq * v0; row0[qq] = a0; r0 += qv * a0;
                float a1 = row1[qq] + kq * v1; row1[qq] = a1; r1 += qv * a1;
            }
            rbf[(size_t)b * DM + head * HDIM + d0]     = (__bf16)r0;
            rbf[(size_t)b * DM + head * HDIM + d0 + 1] = (__bf16)r1;

            if (t == SEQ - 1) {                        // outputs k, q (last step)
                const size_t o = (size_t)b * DM + head * HDIM + d0;
                out[(size_t)BATCH * DM + o]         = ke0;
                out[(size_t)BATCH * DM + o + 1]     = ke1;
                out[(size_t)2 * BATCH * DM + o]     = q0;
                out[(size_t)2 * BATCH * DM + o + 1] = q1;
            }
        }

        grid_barrier(cnt, gen, ++nbar);   // r complete for next step
    }

    // ================= Final: out = r @ Wo.T + bo (WMMA) ===================
    if (blockIdx.x < DM / 16) {                        // 32 WGs, 16 cols each
        const int c0  = blockIdx.x * 16;
        const int col = c0 + nlo;
        v8f acc = {};
        const __bf16* rrow = rbf + (size_t)mrow * DM;
        for (int kk = 0; kk < DM; kk += 32) {
            v16bf a = load_a_frag(rrow, kk, lane);
            v16bf b = load_b_frag(wo, DM, col, kk, lane);
            acc = wmma_bf16(a, b, acc);
        }
        const float bias = bo[col];
        #pragma unroll
        for (int v = 0; v < 8; ++v)
            out[(size_t)(mtop + v) * DM + col] = acc[v] + bias;
    }
}

// ---------------------------------------------------------------------------
// Host entry
// ---------------------------------------------------------------------------
extern "C" void kernel_launch(void* const* d_in, const int* in_sizes, int n_in,
                              void* d_out, int out_size, void* d_ws, size_t ws_size,
                              hipStream_t stream) {
    const float* x   = (const float*)d_in[0];
    const float* Wih = (const float*)d_in[1];
    const float* Whh = (const float*)d_in[2];
    const float* bih = (const float*)d_in[3];
    const float* bhh = (const float*)d_in[4];
    const float* Wo  = (const float*)d_in[5];
    const float* bo  = (const float*)d_in[6];
    char* ws = (char*)d_ws;

    // bf16 weight/input staging (deterministic, redone each launch)
    cvt_bf16_kernel<<<1024, 256, 0, stream>>>(Wih, (__bf16*)(ws + OFF_WIH), GATES * 2 * DM);
    cvt_bf16_kernel<<<1024, 256, 0, stream>>>(Whh, (__bf16*)(ws + OFF_WHH), GATES * HIDN);
    cvt_bf16_kernel<<<256,  256, 0, stream>>>(Wo,  (__bf16*)(ws + OFF_WO),  DM * DM);
    cvt_bf16_kernel<<<2048, 256, 0, stream>>>(x,   (__bf16*)(ws + OFF_X),   SEQ * BATCH * DM);

    // zero h / h_bf16 / r_bf16 / A / sync counters (contiguous region)
    hipMemsetAsync(ws + OFF_ZERO, 0, ZERO_BYTES, stream);

    recurrent_am_kernel<<<NWG, NTHR, 0, stream>>>(ws, bih, bhh, bo, (float*)d_out);
}
